// RingDilatedAttentionV2Robust_73375221285554
// MI455X (gfx1250) — matrix operand order repaired and checked by reading, not compiled, not run on hardware
//
#include <hip/hip_runtime.h>

typedef __attribute__((ext_vector_type(16))) _Float16 v16h;
typedef __attribute__((ext_vector_type(8)))  float    v8f;

#define B_  2
#define N_  2048
#define H_  16
#define D_  64
#define HD  (H_ * D_)

#define WMMA_F16(A, Bm, C) \
  __builtin_amdgcn_wmma_f32_16x16x32_f16(false, (A), false, (Bm), (short)0, (C), false, false)

// Dilation gather: heads 0..7 identity, heads 8..15 -> 1 + 2*(j mod 1024)
__device__ __forceinline__ int map_key(int j, int h) {
  return (h < (H_ / 2)) ? j : (1 + 2 * (j & (N_ / 2 - 1)));
}

// =====================================================================
// Pre-pass: gather + convert K -> Kh[b][h][j][d] (f16, row major) and
// V -> Vt[b][h][d][j] (f16, transposed) so the flash loop does pure
// aligned v16h loads with no per-element conversion or gather math.
// One block = one (b,h,64-row j tile). Reads and writes fully coalesced;
// V transposed through a padded LDS tile.
// =====================================================================
__global__ __launch_bounds__(256)
void prepack_kv(const float* __restrict__ K, const float* __restrict__ V,
                _Float16* __restrict__ Kh, _Float16* __restrict__ Vt)
{
  __shared__ _Float16 T[64][72];          // 64j x 64d, padded row to kill bank conflicts

  const int blk = blockIdx.x;             // B*H*(N/64) = 1024
  const int jt  = blk & (N_ / 64 - 1);
  const int bh  = blk / (N_ / 64);
  const int h   = bh & (H_ - 1);
  const int b   = bh >> 4;
  const int tid = threadIdx.x;

  // phase 1: row-coalesced reads; K straight out, V into LDS
  const int jl = tid >> 2;                // 0..63 local j
  const int dq = (tid & 3) * 16;          // d quarter
  const int j  = jt * 64 + jl;
  const int js = map_key(j, h);

  const float* kin = K + ((size_t)b * N_ + js) * HD + h * D_ + dq;
  const float* vin = V + ((size_t)b * N_ + js) * HD + h * D_ + dq;
  _Float16*   kout = Kh + (((size_t)b * H_ + h) * N_ + j) * D_ + dq;

#pragma unroll
  for (int e = 0; e < 16; ++e) {
    kout[e]       = (_Float16)kin[e];
    T[jl][dq + e] = (_Float16)vin[e];
  }
  __syncthreads();

  // phase 2: write V transposed, coalesced along j
  const int dl = tid >> 2;                // 0..63 d
  const int jq = (tid & 3) * 16;          // j quarter
  _Float16* vout = Vt + (((size_t)b * H_ + h) * D_ + dl) * N_ + jt * 64 + jq;
#pragma unroll
  for (int e = 0; e < 16; ++e) vout[e] = T[jq + e][dl];
}

// =====================================================================
// Fast path: flash attention on pre-packed f16 K/V.
// =====================================================================
__global__ __launch_bounds__(128)
void flash_fast(const float* __restrict__ Q,
                const _Float16* __restrict__ Kh,
                const _Float16* __restrict__ Vt,
                float* __restrict__ O,
                const int* __restrict__ causal_p)
{
  __shared__ _Float16 Plds[4][16][64];    // per-wave P staging (C-layout -> A-layout)

  const int lane = threadIdx.x & 31;
  const int wave = threadIdx.x >> 5;
  const int hi   = lane >> 4;
  const int ln   = lane & 15;

  const int blk = blockIdx.x;             // B*H*(N/64) = 1024
  const int qt  = blk & (N_ / 64 - 1);
  const int bh  = blk / (N_ / 64);
  const int h   = bh & (H_ - 1);
  const int b   = bh >> 4;
  const int q0  = qt * 64 + wave * 16;

  const int   causal = causal_p[0];
  const float scale  = 0.125f;            // 1/sqrt(64)

  const float*    qb  = Q  + (size_t)b * N_ * HD + h * D_;
  const _Float16* khb = Kh + ((size_t)b * H_ + h) * N_ * D_;
  const _Float16* vtb = Vt + ((size_t)b * H_ + h) * D_ * N_;
  float*          ob  = O  + (size_t)b * N_ * HD + h * D_;

  // Q fragments (A layout)
  const float* qrow = qb + (size_t)(q0 + ln) * HD;
  v16h a0, a1;
#pragma unroll
  for (int i = 0; i < 8; ++i) {
    a0[i]     = (_Float16)qrow[hi * 16 + i];
    a0[8 + i] = (_Float16)qrow[hi * 16 + 8 + i];
    a1[i]     = (_Float16)qrow[32 + hi * 16 + i];
    a1[8 + i] = (_Float16)qrow[32 + hi * 16 + 8 + i];
  }

  v16h ones;
#pragma unroll
  for (int i = 0; i < 16; ++i) ones[i] = (_Float16)1.0f;

  v8f acc0 = {}, acc1 = {}, acc2 = {}, acc3 = {};
  v8f accl = {};
  float m_i[8];
#pragma unroll
  for (int r = 0; r < 8; ++r) m_i[r] = -__builtin_inff();

  const int kend = causal ? ((q0 & ~63) + 64) : N_;

  for (int kbk = 0; kbk < kend; kbk += 64) {
    v8f s0 = {}, s1 = {}, s2 = {}, s3 = {};
#define QK_TILE(St, T)                                                      \
    {                                                                       \
      const _Float16* kr = khb + (size_t)(kbk + (T) * 16 + ln) * D_;        \
      v16h bk0 = *(const v16h*)(kr + hi * 16);                              \
      v16h bk1 = *(const v16h*)(kr + 32 + hi * 16);                         \
      St = WMMA_F16(a0, bk0, St);                                           \
      St = WMMA_F16(a1, bk1, St);                                           \
    }
    QK_TILE(s0, 0) QK_TILE(s1, 1) QK_TILE(s2, 2) QK_TILE(s3, 3)
#undef QK_TILE

    float corr[8];
#pragma unroll
    for (int r = 0; r < 8; ++r) {
      const int row = q0 + 8 * hi + r;
      float x0 = s0[r] * scale;
      float x1 = s1[r] * scale;
      float x2 = s2[r] * scale;
      float x3 = s3[r] * scale;
      if (causal) {
        x0 = (kbk + ln      > row) ? -__builtin_inff() : x0;
        x1 = (kbk + 16 + ln > row) ? -__builtin_inff() : x1;
        x2 = (kbk + 32 + ln > row) ? -__builtin_inff() : x2;
        x3 = (kbk + 48 + ln > row) ? -__builtin_inff() : x3;
      }
      float mx = fmaxf(fmaxf(x0, x1), fmaxf(x2, x3));
#pragma unroll
      for (int off = 8; off > 0; off >>= 1)
        mx = fmaxf(mx, __shfl_xor(mx, off, 32));
      const float mn = fmaxf(m_i[r], mx);
      const float p0 = __expf(x0 - mn);
      const float p1 = __expf(x1 - mn);
      const float p2 = __expf(x2 - mn);
      const float p3 = __expf(x3 - mn);
      corr[r] = __expf(m_i[r] - mn);
      m_i[r]  = mn;
      Plds[wave][8 * hi + r][ln]      = (_Float16)p0;
      Plds[wave][8 * hi + r][16 + ln] = (_Float16)p1;
      Plds[wave][8 * hi + r][32 + ln] = (_Float16)p2;
      Plds[wave][8 * hi + r][48 + ln] = (_Float16)p3;
    }

#pragma unroll
    for (int r = 0; r < 8; ++r) {
      acc0[r] *= corr[r]; acc1[r] *= corr[r];
      acc2[r] *= corr[r]; acc3[r] *= corr[r];
      accl[r] *= corr[r];
    }

    asm volatile("s_wait_dscnt 0" ::: "memory");

    v16h pf0, pf1;
#pragma unroll
    for (int e = 0; e < 8; ++e) {
      pf0[e]     = Plds[wave][ln][hi * 16 + e];
      pf0[8 + e] = Plds[wave][ln][hi * 16 + 8 + e];
      pf1[e]     = Plds[wave][ln][32 + hi * 16 + e];
      pf1[8 + e] = Plds[wave][ln][32 + hi * 16 + 8 + e];
    }

    accl = WMMA_F16(pf0, ones, accl);
    accl = WMMA_F16(pf1, ones, accl);

    // V fragments from transposed f16 V: same aligned-v16h pattern as K
#pragma unroll
    for (int dt = 0; dt < 4; ++dt) {
      const _Float16* vr = vtb + (size_t)(dt * 16 + ln) * N_ + kbk;
      v16h bv0 = *(const v16h*)(vr + hi * 16);
      v16h bv1 = *(const v16h*)(vr + 32 + hi * 16);
      v8f& acc = (dt == 0) ? acc0 : (dt == 1) ? acc1 : (dt == 2) ? acc2 : acc3;
      acc = WMMA_F16(pf0, bv0, acc);
      acc = WMMA_F16(pf1, bv1, acc);
    }
  }

#pragma unroll
  for (int r = 0; r < 8; ++r) {
    const float  inv = 1.0f / accl[r];
    const size_t row = (size_t)(q0 + 8 * hi + r) * HD;
    ob[row + 0 * 16 + ln] = acc0[r] * inv;
    ob[row + 1 * 16 + ln] = acc1[r] * inv;
    ob[row + 2 * 16 + ln] = acc2[r] * inv;
    ob[row + 3 * 16 + ln] = acc3[r] * inv;
  }
}

// =====================================================================
// Fallback path (no workspace): direct-from-f32 kernel, STEP-templated
// affine gather.
// =====================================================================
__device__ __forceinline__ v16h load16h(const float* __restrict__ p) {
  v16h r;
#pragma unroll
  for (int i = 0; i < 16; ++i) r[i] = (_Float16)p[i];
  return r;
}

template <int STEP>
__device__ __forceinline__ int jmap(int j) {
  return (STEP == 1) ? j : (1 + 2 * (j & (N_ / 2 - 1)));
}

template <int STEP>
__global__ __launch_bounds__(128)
void flash_direct(const float* __restrict__ Q, const float* __restrict__ K,
                  const float* __restrict__ V, float* __restrict__ O,
                  const int* __restrict__ causal_p)
{
  __shared__ _Float16 Plds[4][16][64];

  const int lane = threadIdx.x & 31;
  const int wave = threadIdx.x >> 5;
  const int hi   = lane >> 4;
  const int ln   = lane & 15;

  const int blk = blockIdx.x;
  const int qt  = blk & (N_ / 64 - 1);
  const int bh  = blk / (N_ / 64);
  const int h   = (bh & 7) + (STEP == 2 ? 8 : 0);
  const int b   = bh >> 3;
  const int q0  = qt * 64 + wave * 16;

  const int   causal = causal_p[0];
  const float scale  = 0.125f;

  const float* qb = Q + (size_t)b * N_ * HD + h * D_;
  const float* kb = K + (size_t)b * N_ * HD + h * D_;
  const float* vb = V + (size_t)b * N_ * HD + h * D_;
  float*       ob = O + (size_t)b * N_ * HD + h * D_;

  const float* qrow = qb + (size_t)(q0 + ln) * HD;
  v16h a0, a1;
#pragma unroll
  for (int i = 0; i < 8; ++i) {
    a0[i]     = (_Float16)qrow[hi * 16 + i];
    a0[8 + i] = (_Float16)qrow[hi * 16 + 8 + i];
    a1[i]     = (_Float16)qrow[32 + hi * 16 + i];
    a1[8 + i] = (_Float16)qrow[32 + hi * 16 + 8 + i];
  }

  v16h ones;
#pragma unroll
  for (int i = 0; i < 16; ++i) ones[i] = (_Float16)1.0f;

  v8f acc0 = {}, acc1 = {}, acc2 = {}, acc3 = {};
  v8f accl = {};
  float m_i[8];
#pragma unroll
  for (int r = 0; r < 8; ++r) m_i[r] = -__builtin_inff();

  const int kend = causal ? ((q0 & ~63) + 64) : N_;

  for (int kbk = 0; kbk < kend; kbk += 64) {
    v8f s0 = {}, s1 = {}, s2 = {}, s3 = {};
#define QK_TILE(St, T)                                                          \
    {                                                                           \
      const float* kr = kb + (size_t)jmap<STEP>(kbk + (T) * 16 + ln) * HD;      \
      v16h bk0 = load16h(kr + hi * 16);                                         \
      v16h bk1 = load16h(kr + 32 + hi * 16);                                    \
      St = WMMA_F16(a0, bk0, St);                                               \
      St = WMMA_F16(a1, bk1, St);                                               \
    }
    QK_TILE(s0, 0) QK_TILE(s1, 1) QK_TILE(s2, 2) QK_TILE(s3, 3)
#undef QK_TILE

    float corr[8];
#pragma unroll
    for (int r = 0; r < 8; ++r) {
      const int row = q0 + 8 * hi + r;
      float x0 = s0[r] * scale;
      float x1 = s1[r] * scale;
      float x2 = s2[r] * scale;
      float x3 = s3[r] * scale;
      if (causal) {
        x0 = (kbk + ln      > row) ? -__builtin_inff() : x0;
        x1 = (kbk + 16 + ln > row) ? -__builtin_inff() : x1;
        x2 = (kbk + 32 + ln > row) ? -__builtin_inff() : x2;
        x3 = (kbk + 48 + ln > row) ? -__builtin_inff() : x3;
      }
      float mx = fmaxf(fmaxf(x0, x1), fmaxf(x2, x3));
#pragma unroll
      for (int off = 8; off > 0; off >>= 1)
        mx = fmaxf(mx, __shfl_xor(mx, off, 32));
      const float mn = fmaxf(m_i[r], mx);
      const float p0 = __expf(x0 - mn);
      const float p1 = __expf(x1 - mn);
      const float p2 = __expf(x2 - mn);
      const float p3 = __expf(x3 - mn);
      corr[r] = __expf(m_i[r] - mn);
      m_i[r]  = mn;
      Plds[wave][8 * hi + r][ln]      = (_Float16)p0;
      Plds[wave][8 * hi + r][16 + ln] = (_Float16)p1;
      Plds[wave][8 * hi + r][32 + ln] = (_Float16)p2;
      Plds[wave][8 * hi + r][48 + ln] = (_Float16)p3;
    }

#pragma unroll
    for (int r = 0; r < 8; ++r) {
      acc0[r] *= corr[r]; acc1[r] *= corr[r];
      acc2[r] *= corr[r]; acc3[r] *= corr[r];
      accl[r] *= corr[r];
    }

    asm volatile("s_wait_dscnt 0" ::: "memory");

    v16h pf0, pf1;
#pragma unroll
    for (int e = 0; e < 8; ++e) {
      pf0[e]     = Plds[wave][ln][hi * 16 + e];
      pf0[8 + e] = Plds[wave][ln][hi * 16 + 8 + e];
      pf1[e]     = Plds[wave][ln][32 + hi * 16 + e];
      pf1[8 + e] = Plds[wave][ln][32 + hi * 16 + 8 + e];
    }

    accl = WMMA_F16(pf0, ones, accl);
    accl = WMMA_F16(pf1, ones, accl);

    const float* rp0 = vb + (size_t)jmap<STEP>(kbk + hi * 16) * HD + ln;
    const float* rp1 = vb + (size_t)jmap<STEP>(kbk + 32 + hi * 16) * HD + ln;
#pragma unroll
    for (int dt = 0; dt < 4; ++dt) {
      v16h bv0, bv1;
#pragma unroll
      for (int e = 0; e < 16; ++e) {
        bv0[e] = (_Float16)rp0[(size_t)(e * STEP * HD + dt * 16)];
        bv1[e] = (_Float16)rp1[(size_t)(e * STEP * HD + dt * 16)];
      }
      v8f& acc = (dt == 0) ? acc0 : (dt == 1) ? acc1 : (dt == 2) ? acc2 : acc3;
      acc = WMMA_F16(pf0, bv0, acc);
      acc = WMMA_F16(pf1, bv1, acc);
    }
  }

#pragma unroll
  for (int r = 0; r < 8; ++r) {
    const float  inv = 1.0f / accl[r];
    const size_t row = (size_t)(q0 + 8 * hi + r) * HD;
    ob[row + 0 * 16 + ln] = acc0[r] * inv;
    ob[row + 1 * 16 + ln] = acc1[r] * inv;
    ob[row + 2 * 16 + ln] = acc2[r] * inv;
    ob[row + 3 * 16 + ln] = acc3[r] * inv;
  }
}

extern "C" void kernel_launch(void* const* d_in, const int* in_sizes, int n_in,
                              void* d_out, int out_size, void* d_ws, size_t ws_size,
                              hipStream_t stream) {
  const float* q  = (const float*)d_in[0];
  const float* k  = (const float*)d_in[1];
  const float* v  = (const float*)d_in[2];
  const int* isc  = (const int*)d_in[3];
  float* out      = (float*)d_out;
  (void)in_sizes; (void)n_in; (void)out_size;

  const size_t elems = (size_t)B_ * H_ * N_ * D_;          // 4,194,304
  const size_t need  = 2 * elems * sizeof(_Float16);       // 16.78 MB

  if (ws_size >= need) {
    _Float16* Kh = (_Float16*)d_ws;
    _Float16* Vt = Kh + elems;
    prepack_kv<<<dim3(B_ * H_ * (N_ / 64)), dim3(256), 0, stream>>>(k, v, Kh, Vt);
    flash_fast<<<dim3(B_ * H_ * (N_ / 64)), dim3(128), 0, stream>>>(q, Kh, Vt, out, isc);
  } else {
    dim3 grid(B_ * (H_ / 2) * (N_ / 64));
    flash_direct<1><<<grid, dim3(128), 0, stream>>>(q, k, v, out, isc);
    flash_direct<2><<<grid, dim3(128), 0, stream>>>(q, k, v, out, isc);
  }
}
